// RVQVAE_66391604461609
// MI455X (gfx1250) — matrix operand order repaired and checked by reading, not compiled, not run on hardware
//
#include <hip/hip_runtime.h>
#include <hip/hip_bf16.h>

typedef __attribute__((ext_vector_type(16))) __bf16 v16bf;
typedef __attribute__((ext_vector_type(8)))  __bf16 v8bf;
typedef __attribute__((ext_vector_type(4)))  __bf16 v4bf;
typedef __attribute__((ext_vector_type(8)))  float  v8f;

#define WMMA_BF16(a, b, c) \
  __builtin_amdgcn_wmma_f32_16x16x32_bf16(false, (a), false, (b), (short)0, (c), false, false)

#define F_RELU_IN  1
#define F_RELU_OUT 2
#define F_ADDRES   4
#define F_UPS      8
#define F_XPOSE    16

// Load a 32-byte fragment (16 bf16) from LDS as two b128 reads.
__device__ __forceinline__ v16bf ld_frag(const __bf16* p) {
  const v8bf* q = (const v8bf*)p;
  v8bf lo = q[0], hi = q[1];
  return __builtin_shufflevector(lo, hi, 0, 1, 2, 3, 4, 5, 6, 7, 8, 9, 10, 11,
                                 12, 13, 14, 15);
}

// ---------------------------------------------------------------------------
// conv1d as implicit GEMM via WMMA bf16 (f32 accumulate), double-buffered LDS.
// Block: 128 threads (4 waves), 64(Cout) x 64(Tout) tile; wave: 32x32.
// LDS tiles stored pre-permuted into WMMA fragment order (ISA 7.12.2):
//   A: As[m*40 + half*16 + e], where k = (e<8?e:e+8)+8*half
//   B: Bs[n*40 + half*16 + e], same k permutation as A
// All global addressing: uniform SGPR base + unsigned 32-bit lane offset.
// ---------------------------------------------------------------------------
template <int KW, int FLAGS>
__global__ __launch_bounds__(128)
void conv1d_wmma(const float* __restrict__ X, const float* __restrict__ W,
                 const float* __restrict__ Bias, const float* __restrict__ Res,
                 float* __restrict__ Y, int Cin, int Cout, int Tin, int Tout,
                 int stride, int pad, int dil) {
  constexpr bool RELU_IN  = (FLAGS & F_RELU_IN)  != 0;
  constexpr bool RELU_OUT = (FLAGS & F_RELU_OUT) != 0;
  constexpr bool ADDRES   = (FLAGS & F_ADDRES)   != 0;
  constexpr bool UPS      = (FLAGS & F_UPS)      != 0;
  constexpr bool XPOSE    = (FLAGS & F_XPOSE)    != 0;

  const int b    = blockIdx.z;
  const int tb   = blockIdx.x * 64;
  const int cb   = blockIdx.y * 64;
  const int tid  = threadIdx.x;
  const int lane = tid & 31;
  const int wv   = tid >> 5;
  const int half = lane >> 4;
  const int l16  = lane & 15;
  const int Ktot = Cin * KW;
  const unsigned uTeff = (unsigned)(UPS ? (Tin * 2) : Tin);
  const bool VECA = ((Ktot & 31) == 0);  // uniform

  // uniform bases (SGPR-resident)
  const float* Xb = X + (size_t)b * Cin * Tin;  // same product either layout
  float* Yb = Y + (size_t)b * Cout * Tout;
  const float* Rb = ADDRES ? (Res + (size_t)b * Cout * Tout) : nullptr;

  __shared__ __attribute__((aligned(16))) __bf16 As2[2][64 * 40];
  __shared__ __attribute__((aligned(16))) __bf16 Bs2[2][64 * 40];

  // loader geometry: 512 quads (4 bf16) per tile, 4 per thread
  int mA[4], e0A[4], hA[4], nB[4], k4B[4], tbB[4];
  unsigned offA[4];  // element offset of quad within W row space
#pragma unroll
  for (int i = 0; i < 4; ++i) {
    int qa = tid + i * 128;
    mA[i] = qa >> 3;
    int k4 = qa & 7;
    hA[i] = (k4 >> 1) & 1;
    e0A[i] = ((k4 & 1) << 2) | ((k4 >> 2) << 3);
    int co = cb + mA[i];
    int coc = co < Cout ? co : (Cout - 1);
    offA[i] = (unsigned)coc * (unsigned)Ktot + (unsigned)(k4 * 4);
    nB[i] = qa & 63;
    k4B[i] = qa >> 6;
    tbB[i] = (tb + nB[i]) * stride - pad;
  }

  v4bf rA[4], rB[4];

  auto stage = [&](int k0) {
#pragma unroll
    for (int i = 0; i < 4; ++i) {  // weights: contiguous along k
      int co = cb + mA[i];
      float f0, f1, f2, f3;
      if (VECA) {
        float4 v = *(const float4*)(W + (offA[i] + (unsigned)k0));
        bool ok = co < Cout;
        f0 = ok ? v.x : 0.f; f1 = ok ? v.y : 0.f;
        f2 = ok ? v.z : 0.f; f3 = ok ? v.w : 0.f;
      } else {
        float f[4];
        unsigned kg0 = (unsigned)k0 + (offA[i] - (unsigned)((cb + mA[i] < Cout
                            ? cb + mA[i] : Cout - 1)) * (unsigned)Ktot);
#pragma unroll
        for (int u = 0; u < 4; ++u) {
          unsigned kg = kg0 + (unsigned)u;
          bool ok = (co < Cout) && (kg < (unsigned)Ktot);
          unsigned kgc = kg < (unsigned)Ktot ? kg : (unsigned)(Ktot - 1);
          unsigned row = offA[i] - kg0 + (unsigned)k0;  // coc*Ktot
          float v = W[row - (unsigned)k0 + kgc];
          f[u] = ok ? v : 0.f;
        }
        f0 = f[0]; f1 = f[1]; f2 = f[2]; f3 = f[3];
      }
      v4bf r;
      r[0] = (__bf16)f0; r[1] = (__bf16)f1;
      r[2] = (__bf16)f2; r[3] = (__bf16)f3;
      rA[i] = r;
    }
#pragma unroll
    for (int i = 0; i < 4; ++i) {  // im2col patch: coalesced along n(=t)
      v4bf r;
#pragma unroll
      for (int u = 0; u < 4; ++u) {
        unsigned kg = (unsigned)(k0 + k4B[i] * 4 + u);
        bool okk = kg < (unsigned)Ktot;
        unsigned kgc = okk ? kg : (unsigned)(Ktot - 1);
        unsigned ci = kgc / (unsigned)KW;        // magic-mul (KW constexpr)
        int kw = (int)(kgc - ci * (unsigned)KW);
        int t = tbB[i] + kw * dil;
        bool okt = (unsigned)t < uTeff;
        unsigned tc = okt ? (unsigned)t : 0u;
        unsigned ts = UPS ? (tc >> 1) : tc;
        unsigned off = XPOSE ? (ts * (unsigned)Cin + ci)
                             : (ci * (unsigned)Tin + ts);
        float v = Xb[off];
        v = (okk && okt) ? v : 0.f;
        if (RELU_IN) v = v > 0.f ? v : 0.f;
        r[u] = (__bf16)v;
      }
      rB[i] = r;
    }
  };

  auto commit = [&](int buf) {
#pragma unroll
    for (int i = 0; i < 4; ++i)
      *(v4bf*)&As2[buf][mA[i] * 40 + hA[i] * 16 + e0A[i]] = rA[i];
#pragma unroll
    for (int i = 0; i < 4; ++i)
      *(v4bf*)&Bs2[buf][nB[i] * 40 + ((k4B[i] >> 1) & 1) * 16 + (((k4B[i] & 1) << 2) | ((k4B[i] >> 2) << 3))] = rB[i];
  };

  v8f acc[2][2];
#pragma unroll
  for (int i = 0; i < 2; ++i)
#pragma unroll
    for (int j = 0; j < 2; ++j)
#pragma unroll
      for (int r = 0; r < 8; ++r) acc[i][j][r] = 0.0f;

  const int wm = (wv >> 1) * 32;
  const int wn = (wv & 1) * 32;
  const int nsteps = (Ktot + 31) >> 5;

  stage(0);
  commit(0);
  __syncthreads();
  for (int s = 0; s < nsteps; ++s) {
    const int cur = s & 1;
    const bool more = (s + 1 < nsteps);
    if (more) stage((s + 1) * 32);  // global loads overlap this step's WMMAs
    const __bf16* Ab = As2[cur];
    const __bf16* Bb = Bs2[cur];
    v16bf a0 = ld_frag(&Ab[(wm + l16) * 40 + half * 16]);
    v16bf a1 = ld_frag(&Ab[(wm + 16 + l16) * 40 + half * 16]);
    v16bf b0 = ld_frag(&Bb[(wn + l16) * 40 + half * 16]);
    v16bf b1 = ld_frag(&Bb[(wn + 16 + l16) * 40 + half * 16]);
    acc[0][0] = WMMA_BF16(a0, b0, acc[0][0]);
    acc[0][1] = WMMA_BF16(a0, b1, acc[0][1]);
    acc[1][0] = WMMA_BF16(a1, b0, acc[1][0]);
    acc[1][1] = WMMA_BF16(a1, b1, acc[1][1]);
    if (more) commit(cur ^ 1);
    __syncthreads();
  }

  // epilogue: C/D layout, lane L: N = l16, VGPR r: M = r + 8*half
#pragma unroll
  for (int sm = 0; sm < 2; ++sm)
#pragma unroll
    for (int sn = 0; sn < 2; ++sn)
#pragma unroll
      for (int r = 0; r < 8; ++r) {
        int m = cb + wm + sm * 16 + r + half * 8;
        int n = tb + wn + sn * 16 + l16;
        if (m < Cout) {
          unsigned off = (unsigned)m * (unsigned)Tout + (unsigned)n;
          float v = acc[sm][sn][r] + Bias[m];
          if (ADDRES) v += Rb[off];
          if (RELU_OUT && v < 0.f) v = 0.f;
          Yb[off] = v;
        }
      }
}

// ---------------------------------------------------------------------------
// RVQ score GEMM: D[n,c] = cbsq[q,c] - 2 * residual[n,:] . cb[q,c,:]
// residual [32,512,64] (b,d,t), n = b*64+t.  M=2048, N=1024, K=512.
// ---------------------------------------------------------------------------
__global__ __launch_bounds__(128)
void rvq_score(const float* __restrict__ Rres, const float* __restrict__ CB,
               const float* __restrict__ cbsq, float* __restrict__ Dout,
               int q) {
  const int nb    = blockIdx.y * 64;
  const int cbase = blockIdx.x * 64;
  const int bidx  = blockIdx.y;  // batch: n tiles align with batches
  const int tid  = threadIdx.x;
  const int lane = tid & 31;
  const int wv   = tid >> 5;
  const int half = lane >> 4;
  const int l16  = lane & 15;

  // uniform bases
  const float* Rb = Rres + (size_t)bidx * 32768;                  // [512][64]
  const float* Cb = CB + (((size_t)q << 10) + (size_t)cbase) * 512;
  float* Db = Dout + (size_t)nb * 1024 + cbase;

  __shared__ __attribute__((aligned(16))) __bf16 As2[2][64 * 40];
  __shared__ __attribute__((aligned(16))) __bf16 Bs2[2][64 * 40];

  int mA[4], e0A[4], hA[4], nBv[4], k4B[4];
  unsigned offA0[4], offB0[4];
#pragma unroll
  for (int i = 0; i < 4; ++i) {
    int qa = tid + i * 128;
    mA[i] = qa & 63;  // t-local, coalesced
    int k4 = qa >> 6;
    hA[i] = (k4 >> 1) & 1;
    e0A[i] = ((k4 & 1) << 2) | ((k4 >> 2) << 3);
    offA0[i] = (unsigned)(k4 * 4 * 64 + mA[i]);  // d-major stride 64
    k4B[i] = qa & 7;  // coalesced float4 chunks along d
    nBv[i] = qa >> 3;
    offB0[i] = (unsigned)(nBv[i] * 512 + k4B[i] * 4);
  }

  v4bf rA[4], rB[4];

  auto stage = [&](int k0) {
#pragma unroll
    for (int i = 0; i < 4; ++i) {
      unsigned o = offA0[i] + (unsigned)k0 * 64u;
      v4bf r;
#pragma unroll
      for (int u = 0; u < 4; ++u)
        r[u] = (__bf16)Rb[o + (unsigned)(u * 64)];  // 256B immediate strides
      rA[i] = r;
    }
#pragma unroll
    for (int i = 0; i < 4; ++i) {
      float4 v = *(const float4*)(Cb + (offB0[i] + (unsigned)k0));
      v4bf r;
      r[0] = (__bf16)v.x; r[1] = (__bf16)v.y;
      r[2] = (__bf16)v.z; r[3] = (__bf16)v.w;
      rB[i] = r;
    }
  };

  auto commit = [&](int buf) {
#pragma unroll
    for (int i = 0; i < 4; ++i)
      *(v4bf*)&As2[buf][mA[i] * 40 + hA[i] * 16 + e0A[i]] = rA[i];
#pragma unroll
    for (int i = 0; i < 4; ++i)
      *(v4bf*)&Bs2[buf][nBv[i] * 40 + ((k4B[i] >> 1) & 1) * 16 + (((k4B[i] & 1) << 2) | ((k4B[i] >> 2) << 3))] = rB[i];
  };

  v8f acc[2][2];
#pragma unroll
  for (int i = 0; i < 2; ++i)
#pragma unroll
    for (int j = 0; j < 2; ++j)
#pragma unroll
      for (int r = 0; r < 8; ++r) acc[i][j][r] = 0.0f;

  const int wm = (wv >> 1) * 32;
  const int wn = (wv & 1) * 32;

  stage(0);
  commit(0);
  __syncthreads();
  for (int s = 0; s < 16; ++s) {
    const int cur = s & 1;
    const bool more = (s + 1 < 16);
    if (more) stage((s + 1) * 32);
    const __bf16* Ab = As2[cur];
    const __bf16* Bb = Bs2[cur];
    v16bf a0 = ld_frag(&Ab[(wm + l16) * 40 + half * 16]);
    v16bf a1 = ld_frag(&Ab[(wm + 16 + l16) * 40 + half * 16]);
    v16bf b0 = ld_frag(&Bb[(wn + l16) * 40 + half * 16]);
    v16bf b1 = ld_frag(&Bb[(wn + 16 + l16) * 40 + half * 16]);
    acc[0][0] = WMMA_BF16(a0, b0, acc[0][0]);
    acc[0][1] = WMMA_BF16(a0, b1, acc[0][1]);
    acc[1][0] = WMMA_BF16(a1, b0, acc[1][0]);
    acc[1][1] = WMMA_BF16(a1, b1, acc[1][1]);
    if (more) commit(cur ^ 1);
    __syncthreads();
  }

  const float* csq = cbsq + (q << 10) + cbase;
#pragma unroll
  for (int sm = 0; sm < 2; ++sm)
#pragma unroll
    for (int sn = 0; sn < 2; ++sn)
#pragma unroll
      for (int r = 0; r < 8; ++r) {
        unsigned nl = (unsigned)(wm + sm * 16 + r + half * 8);
        unsigned cl = (unsigned)(wn + sn * 16 + l16);
        Db[nl * 1024u + cl] = csq[cl] - 2.0f * acc[sm][sn][r];
      }
}

// ---------------------------------------------------------------------------
__global__ __launch_bounds__(256)
void rvq_argmin(const float* __restrict__ D, int* __restrict__ idx) {
  const int n = blockIdx.x;
  __shared__ float sv[256];
  __shared__ int si[256];
  float best = 3.4e38f;
  int bi = 0x7fffffff;
  for (int c = threadIdx.x; c < 1024; c += 256) {
    float v = D[(size_t)n * 1024 + c];
    if (v < best || (v == best && c < bi)) { best = v; bi = c; }
  }
  sv[threadIdx.x] = best; si[threadIdx.x] = bi;
  __syncthreads();
  for (int s = 128; s > 0; s >>= 1) {
    if ((int)threadIdx.x < s) {
      float ov = sv[threadIdx.x + s]; int oi = si[threadIdx.x + s];
      if (ov < sv[threadIdx.x] ||
          (ov == sv[threadIdx.x] && oi < si[threadIdx.x])) {
        sv[threadIdx.x] = ov; si[threadIdx.x] = oi;
      }
    }
    __syncthreads();
  }
  if (threadIdx.x == 0) idx[n] = si[0];
}

__global__ void rvq_hist(const int* __restrict__ idx, int* __restrict__ hist) {
  int n = blockIdx.x * blockDim.x + threadIdx.x;
  if (n < 2048) atomicAdd(&hist[idx[n]], 1);
}

__global__ __launch_bounds__(256)
void rvq_perp(const int* __restrict__ hist, float* __restrict__ perps, int q) {
  __shared__ float s[256];
  float a = 0.0f;
  for (int c = threadIdx.x; c < 1024; c += 256) {
    float p = (float)hist[c] * (1.0f / 2048.0f);
    if (p > 0.0f) a += p * logf(p);
  }
  s[threadIdx.x] = a;
  __syncthreads();
  for (int st = 128; st > 0; st >>= 1) {
    if ((int)threadIdx.x < st) s[threadIdx.x] += s[threadIdx.x + st];
    __syncthreads();
  }
  if (threadIdx.x == 0) perps[q] = expf(-s[0]);
}

__global__ __launch_bounds__(64)
void rvq_update(float* __restrict__ Rres, float* __restrict__ Q,
                const float* __restrict__ CB, const int* __restrict__ idx,
                float* __restrict__ lossp, int q) {
  const int blk = blockIdx.x;  // (b, d)
  const int b = blk >> 9;
  const int d = blk & 511;
  const int t = threadIdx.x;
  const int code = idx[b * 64 + t];
  const float xd = CB[(((size_t)q << 10) + code) * 512 + d];
  const size_t a = (((size_t)b * 512) + d) * 64 + t;
  float rn = Rres[a] - xd;
  Rres[a] = rn;
  Q[a] += xd;
  __shared__ float s[64];
  s[t] = rn * rn;
  __syncthreads();
  for (int st = 32; st > 0; st >>= 1) {
    if (t < st) s[t] += s[t + st];
    __syncthreads();
  }
  if (t == 0) lossp[blk] = s[0];
}

__global__ __launch_bounds__(256)
void reduce_sum(const float* __restrict__ in, int n, float* __restrict__ out) {
  __shared__ float s[256];
  float a = 0.0f;
  for (int i = threadIdx.x; i < n; i += 256) a += in[i];
  s[threadIdx.x] = a;
  __syncthreads();
  for (int st = 128; st > 0; st >>= 1) {
    if ((int)threadIdx.x < st) s[threadIdx.x] += s[threadIdx.x + st];
    __syncthreads();
  }
  if (threadIdx.x == 0) out[0] = s[0];
}

__global__ __launch_bounds__(128)
void cbsq_kernel(const float* __restrict__ CB, float* __restrict__ cbsq) {
  const int row = blockIdx.x;
  __shared__ float s[128];
  float a = 0.0f;
  for (int d = threadIdx.x; d < 512; d += 128) {
    float v = CB[(size_t)row * 512 + d];
    a += v * v;
  }
  s[threadIdx.x] = a;
  __syncthreads();
  for (int st = 64; st > 0; st >>= 1) {
    if ((int)threadIdx.x < st) s[threadIdx.x] += s[threadIdx.x + st];
    __syncthreads();
  }
  if (threadIdx.x == 0) cbsq[row] = s[0];
}

__global__ void fill_kernel(float* __restrict__ p, float v, int n) {
  int i = blockIdx.x * blockDim.x + threadIdx.x;
  if (i < n) p[i] = v;
}

__global__ void rvq_final(const float* __restrict__ losses,
                          const float* __restrict__ perps,
                          float* __restrict__ out) {
  if (threadIdx.x == 0 && blockIdx.x == 0) {
    float cl = 0.0f, pp = 0.0f;
    for (int q = 0; q < 6; ++q) { cl += losses[q]; pp += perps[q]; }
    out[0] = cl * (1.0f / 1048576.0f);
    out[1] = pp * (1.0f / 6.0f);
  }
}

// ---------------------------------------------------------------------------
extern "C" void kernel_launch(void* const* d_in, const int* in_sizes, int n_in,
                              void* d_out, int out_size, void* d_ws,
                              size_t ws_size, hipStream_t stream) {
  int cu = 0;
  const float* x = (const float*)d_in[cu++];
  const float* e_in_w  = (const float*)d_in[cu++];
  const float* e_in_b  = (const float*)d_in[cu++];
  const float* e_out_w = (const float*)d_in[cu++];
  const float* e_out_b = (const float*)d_in[cu++];
  const float *e_down_w[3], *e_down_b[3];
  const float *e_w1[3][3], *e_b1[3][3], *e_w2[3][3], *e_b2[3][3];
  for (int i = 0; i < 3; ++i) {
    e_down_w[i] = (const float*)d_in[cu++];
    e_down_b[i] = (const float*)d_in[cu++];
    for (int j = 0; j < 3; ++j) {
      e_w1[i][j] = (const float*)d_in[cu++];
      e_b1[i][j] = (const float*)d_in[cu++];
      e_w2[i][j] = (const float*)d_in[cu++];
      e_b2[i][j] = (const float*)d_in[cu++];
    }
  }
  const float* de_in_w  = (const float*)d_in[cu++];
  const float* de_in_b  = (const float*)d_in[cu++];
  const float* de_mid_w = (const float*)d_in[cu++];
  const float* de_mid_b = (const float*)d_in[cu++];
  const float* de_out_w = (const float*)d_in[cu++];
  const float* de_out_b = (const float*)d_in[cu++];
  const float *de_up_w[3], *de_up_b[3];
  const float *de_w1[3][3], *de_b1[3][3], *de_w2[3][3], *de_b2[3][3];
  for (int i = 0; i < 3; ++i) {
    de_up_w[i] = (const float*)d_in[cu++];
    de_up_b[i] = (const float*)d_in[cu++];
    for (int j = 0; j < 3; ++j) {
      de_w1[i][j] = (const float*)d_in[cu++];
      de_b1[i][j] = (const float*)d_in[cu++];
      de_w2[i][j] = (const float*)d_in[cu++];
      de_b2[i][j] = (const float*)d_in[cu++];
    }
  }
  const float* CB = (const float*)d_in[cu++];  // [6,1024,512]

  float* ws = (float*)d_ws;
  const size_t BIG = (size_t)32 * 512 * 512;
  const size_t SML = (size_t)32 * 512 * 64;
  float* X0 = ws;
  float* X1 = X0 + BIG;
  float* X2 = X1 + BIG;
  float* R  = X2 + BIG;
  float* Q  = R + SML;
  float* S  = Q + SML;
  float* CBSQ  = S + (size_t)2048 * 1024;
  float* LOSS  = CBSQ + 6144;
  float* PERP  = LOSS + 8;
  float* LOSSP = PERP + 8;
  int*   IDX   = (int*)(LOSSP + 16384);
  int*   HIST  = IDX + 2048;

  auto conv = [&](const float* Xp, const float* Wp, const float* Bp,
                  const float* Resp, float* Yp, int Cin, int Cout, int Tin,
                  int Tout, int KW, int stride, int pad, int dil, int flags) {
    dim3 g(Tout / 64, (Cout + 63) / 64, 32), blk(128);
    switch (KW * 100 + flags) {
      case 300 + (F_RELU_OUT | F_XPOSE):
        conv1d_wmma<3, F_RELU_OUT | F_XPOSE><<<g, blk, 0, stream>>>(
            Xp, Wp, Bp, Resp, Yp, Cin, Cout, Tin, Tout, stride, pad, dil);
        break;
      case 400:
        conv1d_wmma<4, 0><<<g, blk, 0, stream>>>(Xp, Wp, Bp, Resp, Yp, Cin,
                                                 Cout, Tin, Tout, stride, pad,
                                                 dil);
        break;
      case 300 + F_RELU_IN:
        conv1d_wmma<3, F_RELU_IN><<<g, blk, 0, stream>>>(
            Xp, Wp, Bp, Resp, Yp, Cin, Cout, Tin, Tout, stride, pad, dil);
        break;
      case 100 + (F_RELU_IN | F_ADDRES):
        conv1d_wmma<1, F_RELU_IN | F_ADDRES><<<g, blk, 0, stream>>>(
            Xp, Wp, Bp, Resp, Yp, Cin, Cout, Tin, Tout, stride, pad, dil);
        break;
      case 300:
        conv1d_wmma<3, 0><<<g, blk, 0, stream>>>(Xp, Wp, Bp, Resp, Yp, Cin,
                                                 Cout, Tin, Tout, stride, pad,
                                                 dil);
        break;
      case 300 + F_RELU_OUT:
        conv1d_wmma<3, F_RELU_OUT><<<g, blk, 0, stream>>>(
            Xp, Wp, Bp, Resp, Yp, Cin, Cout, Tin, Tout, stride, pad, dil);
        break;
      case 300 + F_UPS:
        conv1d_wmma<3, F_UPS><<<g, blk, 0, stream>>>(
            Xp, Wp, Bp, Resp, Yp, Cin, Cout, Tin, Tout, stride, pad, dil);
        break;
      default: break;
    }
  };

  // ---------------- encoder ----------------
  conv(x, e_in_w, e_in_b, nullptr, X0, 263, 512, 512, 512, 3, 1, 1, 1,
       F_RELU_OUT | F_XPOSE);
  float *h = X0, *t1 = X1, *t2 = X2;
  int T = 512;
  for (int i = 0; i < 3; ++i) {
    conv(h, e_down_w[i], e_down_b[i], nullptr, t1, 512, 512, T, T / 2, 4, 2, 1,
         1, 0);
    { float* tmp = h; h = t1; t1 = tmp; }
    T /= 2;
    int dr = 1;
    for (int j = 0; j < 3; ++j) {
      conv(h, e_w1[i][j], e_b1[i][j], nullptr, t1, 512, 512, T, T, 3, 1, dr,
           dr, F_RELU_IN);
      conv(t1, e_w2[i][j], e_b2[i][j], h, t2, 512, 512, T, T, 1, 1, 0, 1,
           F_RELU_IN | F_ADDRES);
      { float* tmp = h; h = t2; t2 = tmp; }
      dr *= 3;
    }
  }
  conv(h, e_out_w, e_out_b, nullptr, R, 512, 512, 64, 64, 3, 1, 1, 1, 0);

  // ---------------- RVQ ----------------
  fill_kernel<<<(int)((SML + 255) / 256), 256, 0, stream>>>(Q, 0.0f, (int)SML);
  cbsq_kernel<<<6144, 128, 0, stream>>>(CB, CBSQ);
  for (int q = 0; q < 6; ++q) {
    rvq_score<<<dim3(16, 32), 128, 0, stream>>>(R, CB, CBSQ, S, q);
    rvq_argmin<<<2048, 256, 0, stream>>>(S, IDX);
    fill_kernel<<<4, 256, 0, stream>>>((float*)HIST, 0.0f, 1024);
    rvq_hist<<<8, 256, 0, stream>>>(IDX, HIST);
    rvq_perp<<<1, 256, 0, stream>>>(HIST, PERP, q);
    rvq_update<<<16384, 64, 0, stream>>>(R, Q, CB, IDX, LOSSP, q);
    reduce_sum<<<1, 256, 0, stream>>>(LOSSP, 16384, LOSS + q);
  }

  // ---------------- decoder ----------------
  conv(Q, de_in_w, de_in_b, nullptr, X0, 512, 512, 64, 64, 3, 1, 1, 1,
       F_RELU_OUT);
  h = X0; t1 = X1; t2 = X2;
  T = 64;
  for (int i = 0; i < 3; ++i) {
    int dr = 9;
    for (int j = 0; j < 3; ++j) {
      conv(h, de_w1[i][j], de_b1[i][j], nullptr, t1, 512, 512, T, T, 3, 1, dr,
           dr, F_RELU_IN);
      conv(t1, de_w2[i][j], de_b2[i][j], h, t2, 512, 512, T, T, 1, 1, 0, 1,
           F_RELU_IN | F_ADDRES);
      { float* tmp = h; h = t2; t2 = tmp; }
      dr /= 3;
    }
    conv(h, de_up_w[i], de_up_b[i], nullptr, t1, 512, 512, T, 2 * T, 3, 1, 1,
         1, F_UPS);
    { float* tmp = h; h = t1; t1 = tmp; }
    T *= 2;
  }
  conv(h, de_mid_w, de_mid_b, nullptr, t1, 512, 512, 512, 512, 3, 1, 1, 1,
       F_RELU_OUT);
  { float* tmp = h; h = t1; t1 = tmp; }
  conv(h, de_out_w, de_out_b, nullptr, (float*)d_out, 512, 263, 512, 512, 3, 1,
       1, 1, 0);

  rvq_final<<<1, 32, 0, stream>>>(LOSS, PERP,
                                  (float*)d_out + (size_t)32 * 263 * 512);
}
